// FCBlock_24575802868020
// MI455X (gfx1250) — matrix operand order repaired and checked by reading, not compiled, not run on hardware
//
#include <hip/hip_runtime.h>

// ---------------------------------------------------------------------------
// FCBlock fused forward for MI455X (gfx1250, wave32, WMMA + async LDS staging)
// N=65536 points, H=512, L=3. Output: (N,1) fp32.
// ---------------------------------------------------------------------------

typedef __bf16 bf16_t;
typedef __attribute__((ext_vector_type(16))) __bf16 v16bf;
typedef __attribute__((ext_vector_type(8)))  __bf16 v8bf;
typedef __attribute__((ext_vector_type(8)))  float  v8f;
typedef __attribute__((ext_vector_type(4)))  int    v4i;

#define LDS_AS __attribute__((address_space(3)))
#define GLB_AS __attribute__((address_space(1)))

#define HDIM   512
#define HH     (512 * 512)
#define LDSS   520            // act-buffer row stride (bf16) -> conflict-free A loads
#define PSTRIDE 40            // weight-panel row stride (bf16): 20 dwords -> all 64 banks
#define MROWS  64             // rows per block
#define NPTS   65536

// LDS layout (bytes)
#define BUF_BYTES   (MROWS * LDSS * 2)           // 66560 per activation buffer
#define OFF_BUF0    0
#define OFF_BUF1    (BUF_BYTES)
#define OFF_BUF2    (2 * BUF_BYTES)
#define OFF_YV      (3 * BUF_BYTES)              // 64 floats
#define OFF_T3C     (OFF_YV + 64 * 4)            // 64 floats
#define OFF_RED     (OFF_T3C + 64 * 4)           // 768 floats
#define OFF_CXY     (OFF_RED + 768 * 4)          // 256 floats
#define OFF_PAN0    (OFF_CXY + 256 * 4)          // 204288
#define PAN_BYTES   (HDIM * PSTRIDE * 2)         // 40960: 512 rows x 32 K bf16 (+pad)
#define OFF_PAN1    (OFF_PAN0 + PAN_BYTES)
#define SMEM_BYTES  (OFF_PAN1 + PAN_BYTES)       // 286208 <= 320KB/WGP

// ---- CDNA5 async global->LDS copy (guarded; fallback = register relay) -----
#if defined(__AMDGCN__) && __has_builtin(__builtin_amdgcn_global_load_async_to_lds_b128)
#define HAVE_ASYNC 1
#else
#define HAVE_ASYNC 0
#endif

__device__ __forceinline__ void async_cp16(const bf16_t* gp, bf16_t* lp) {
#if HAVE_ASYNC
  __builtin_amdgcn_global_load_async_to_lds_b128(
      (GLB_AS v4i*)gp, (LDS_AS v4i*)lp, 0, 0);
#else
  *(LDS_AS v8bf*)lp = *(const GLB_AS v8bf*)gp;
#endif
}

__device__ __forceinline__ void async_wait_all() {
#if HAVE_ASYNC
#if __has_builtin(__builtin_amdgcn_s_wait_asynccnt)
  __builtin_amdgcn_s_wait_asynccnt(0);
#else
  asm volatile("s_wait_asynccnt 0x0" ::: "memory");
#endif
#endif
}

struct Params {
  const float* coords;
  const float* Wu0;       // (3,H)
  const float* bu0;       // (H)
  const float* bu;        // (2,H)
  const float* bzuu;      // (3,H)
  const float* Wzzu_last; // (H)
  const float* Wyuu0;     // (3)
  const float* byuu0;     // (1)
  const float* Wyuu;      // (3,H)
  const float* byuu;      // (3)
  const float* Wzyu;      // (3,H)
  const float* Wzyu_last; // (1)
  const float* Wzu0;      // (3,H)
  const float* bzu0;      // (H)
  const float* bzu;       // (2,H)
  const float* Wzu_last;  // (H)
  const float* bzu_last;  // (1)
  const float* p;
  const float* pu;
  const float* pzu;
  const bf16_t* ws;       // 9 transposed bf16 HxH weights
  float* out;
};

// --- prologue: fp32 HxH -> bf16 transposed (Wt[n*H+k] = W[k*H+n]) ----------
__global__ void convert_transpose(const float* __restrict__ src,
                                  bf16_t* __restrict__ dst) {
  __shared__ float tile[32][33];
  const int bx = blockIdx.x, by = blockIdx.y;
  const int tx = threadIdx.x, ty = threadIdx.y;
  for (int j = ty; j < 32; j += 8)
    tile[j][tx] = src[(by * 32 + j) * HDIM + bx * 32 + tx];
  __syncthreads();
  for (int j = ty; j < 32; j += 8)
    dst[(bx * 32 + j) * HDIM + by * 32 + tx] = (bf16_t)tile[tx][j];
}

__device__ __forceinline__ v16bf combine(v8bf lo, v8bf hi) {
  return __builtin_shufflevector(lo, hi, 0,1,2,3,4,5,6,7,8,9,10,11,12,13,14,15);
}

// issue one 32Kx512N panel (32KB) of Wt column-slab kt: 8x b128 per thread
__device__ __forceinline__ void issue_panel(const bf16_t* wt, int kt,
                                            bf16_t* panel, int tid) {
#pragma unroll
  for (int t = 0; t < 8; ++t) {
    const int idx = t * 256 + tid;
    const int n = idx >> 2, c = idx & 3;
    async_cp16(wt + n * HDIM + kt * 32 + c * 8, panel + n * PSTRIDE + c * 8);
  }
}

// one kt-step: 16 WMMAs from one A fragment; B fragments addressed off ONE
// base VGPR with constant DS offsets (j*1280B), software-pipelined one ahead.
__device__ __forceinline__ void wmma_step(v8f acc[16], const bf16_t* albuf,
                                          const bf16_t* pan, int mt, int nh,
                                          int kt, int lane) {
  const int r   = mt * 16 + (lane & 15);
  const int klo = kt * 32 + ((lane >> 4) << 3);
  const LDS_AS v8bf* abase = (const LDS_AS v8bf*)(albuf + r * LDSS + klo);
  v16bf A = combine(abase[0], abase[2]);          // +16 elems == +2 v8bf

  const int col = lane & 15;
  const int kb  = (lane >> 4) << 3;
  const LDS_AS v8bf* bbase =
      (const LDS_AS v8bf*)(pan + (nh * 256 + col) * PSTRIDE + kb);
  // fragment j lives at bbase[80*j], bbase[80*j+2]  (80 v8bf = 640 elems = j stride)
  v16bf Bcur = combine(bbase[0], bbase[2]);
#pragma unroll
  for (int j = 0; j < 16; ++j) {
    v16bf Bnxt = Bcur;
    if (j < 15) Bnxt = combine(bbase[80 * (j + 1)], bbase[80 * (j + 1) + 2]);
    acc[j] = __builtin_amdgcn_wmma_f32_16x16x32_bf16(
        false, A, false, Bcur, (short)0, acc[j], false, false);
    Bcur = Bnxt;
  }
}

// Pipelined K-sweep with statically addressed double buffer (no pointer selects).
// Must be called by all 256 threads (contains __syncthreads).
__device__ __forceinline__ void gemm_async_pass(v8f acc[16], const bf16_t* albuf,
                                                const bf16_t* wt,
                                                bf16_t* P0, bf16_t* P1,
                                                int mt, int nh, int lane, int tid) {
  issue_panel(wt, 0, P0, tid);
  async_wait_all();
  __syncthreads();                         // P0 ready for everyone
  issue_panel(wt, 1, P1, tid);
#pragma unroll 1
  for (int kt = 0; kt < 16; kt += 2) {
    // even step: compute on P0, refill P0 with kt+2
    wmma_step(acc, albuf, P0, mt, nh, kt, lane);
    async_wait_all();                      // my loads for panel kt+1 (in P1) done
    __syncthreads();                       // ready(kt+1) + release(kt)
    if (kt + 2 < 16) issue_panel(wt, kt + 2, P0, tid);
    // odd step: compute on P1, refill P1 with kt+3
    wmma_step(acc, albuf, P1, mt, nh, kt + 1, lane);
    if (kt + 1 < 15) async_wait_all();     // my loads for panel kt+2 (in P0) done
    __syncthreads();                       // ready(kt+2) + release(kt+1)
    if (kt + 3 < 16) issue_panel(wt, kt + 3, P1, tid);
  }
}

// --- main fused kernel ------------------------------------------------------
__global__ __launch_bounds__(256) void fcblock_fused(Params pr) {
  extern __shared__ __align__(16) char smem[];
  bf16_t* buf0 = reinterpret_cast<bf16_t*>(smem + OFF_BUF0);
  bf16_t* buf1 = reinterpret_cast<bf16_t*>(smem + OFF_BUF1);
  bf16_t* buf2 = reinterpret_cast<bf16_t*>(smem + OFF_BUF2);
  float*  yv   = reinterpret_cast<float*>(smem + OFF_YV);
  float*  t3c  = reinterpret_cast<float*>(smem + OFF_T3C);
  float*  red  = reinterpret_cast<float*>(smem + OFF_RED);
  float*  cxy  = reinterpret_cast<float*>(smem + OFF_CXY);
  bf16_t* pan0 = reinterpret_cast<bf16_t*>(smem + OFF_PAN0);
  bf16_t* pan1 = reinterpret_cast<bf16_t*>(smem + OFF_PAN1);

#define LF(p)  (*(LDS_AS float*)&(p))            // LDS float lvalue
#define LB(p)  (*(LDS_AS bf16_t*)&(p))           // LDS bf16 lvalue

  const int tid  = threadIdx.x;
  const int lane = tid & 31;
  const int wid  = tid >> 5;
  const int mt   = wid & 3;   // 16-row tile owned by this wave
  const int nh   = wid >> 2;  // which half of the 512-wide N range
  const int gr0  = blockIdx.x * MROWS;

  const float p10   = 10.0f * pr.p[0];
  const float pu10  = 10.0f * pr.pu[0];
  const float pzu10 = 10.0f * pr.pzu[0];
  const float wy00 = pr.Wyuu0[0], wy01 = pr.Wyuu0[1], wy02 = pr.Wyuu0[2];
  const float by0  = pr.byuu0[0];

  // stage coords for this block's 64 rows
  LF(cxy[tid]) = pr.coords[gr0 * 4 + tid];
  __syncthreads();

  // ---- init layer: z0, u0 (fp32 math, bf16 store) ----
  bf16_t* U = buf0;
  bf16_t* Z = buf1;
  bf16_t* T = buf2;
  for (int idx = tid; idx < MROWS * HDIM; idx += 256) {
    const int r = idx >> 9, h = idx & 511;
    const float c0 = LF(cxy[4 * r]), c1 = LF(cxy[4 * r + 1]);
    const float c2 = LF(cxy[4 * r + 2]), y = LF(cxy[4 * r + 3]);
    const float s  = c0 * wy00 + c1 * wy01 + c2 * wy02 + by0;
    const float zp = c0 * pr.Wzu0[h] + c1 * pr.Wzu0[HDIM + h] + c2 * pr.Wzu0[2 * HDIM + h]
                   + pr.bzu0[h] + y * s * pr.Wzyu[h];
    const float up = c0 * pr.Wu0[h] + c1 * pr.Wu0[HDIM + h] + c2 * pr.Wu0[2 * HDIM + h]
                   + pr.bu0[h];
    LB(Z[r * LDSS + h]) = (bf16_t)fmaxf(p10 * zp, 0.0f);
    LB(U[r * LDSS + h]) = (bf16_t)fmaxf(pu10 * up, 0.0f);
  }
  if (tid < 64) LF(yv[tid]) = LF(cxy[4 * tid + 3]);
  __syncthreads();

  const int col = lane & 15;
  const int rb  = mt * 16 + ((lane >> 4) << 3);

  // ---- layers 1..3 ----
  for (int i = 1; i <= 3; ++i) {
    // GEMM1: a = u @ Wzuu[i-1] + bzuu[i-1]; zz = z * relu(10*pzu*a) -> T
    {
      v8f acc[16];
#pragma unroll
      for (int j = 0; j < 16; ++j) acc[j] = (v8f)(0.0f);
      gemm_async_pass(acc, U, pr.ws + (i - 1) * HH, pan0, pan1, mt, nh, lane, tid);
      const float* bzuu_i = pr.bzuu + (i - 1) * HDIM;
#pragma unroll
      for (int j = 0; j < 16; ++j) {
        const int n  = (nh * 16 + j) * 16 + col;
        const float bz = bzuu_i[n];
#pragma unroll
        for (int e = 0; e < 8; ++e) {
          const int r = rb + e;
          const float a  = acc[j][e] + bz;
          const float zv = (float)LB(Z[r * LDSS + n]);
          LB(T[r * LDSS + n]) = (bf16_t)(zv * fmaxf(pzu10 * a, 0.0f));
        }
      }
    }
    __syncthreads();

    if (i == 3) break;  // T holds zz, U holds u -> final reduction below

    // row scalar: s_r = u_r . Wyuu[i-1] + byuu[i-1]; t3c[r] = y_r * s_r
    {
      const int r = tid >> 2, q = tid & 3, h0 = q * 128;
      const float* wy = pr.Wyuu + (i - 1) * HDIM;
      float s = 0.0f;
      for (int h = h0; h < h0 + 128; ++h) s += (float)LB(U[r * LDSS + h]) * wy[h];
      LF(red[tid]) = s;
    }
    __syncthreads();
    if (tid < 64) {
      const float s = LF(red[4 * tid]) + LF(red[4 * tid + 1]) + LF(red[4 * tid + 2])
                    + LF(red[4 * tid + 3]) + pr.byuu[i - 1];
      LF(t3c[tid]) = LF(yv[tid]) * s;
    }
    __syncthreads();

    // GEMM2+3: t = zz @ Wzzu[i-1] + u @ Wzu[i-1] + bzu[i-1] + t3c*Wzyu[i]
    //          new z = relu(10*p*t) -> Z
    {
      v8f acc[16];
#pragma unroll
      for (int j = 0; j < 16; ++j) acc[j] = (v8f)(0.0f);
      gemm_async_pass(acc, T, pr.ws + (3 + (i - 1)) * HH, pan0, pan1, mt, nh, lane, tid);
      gemm_async_pass(acc, U, pr.ws + (5 + (i - 1)) * HH, pan0, pan1, mt, nh, lane, tid);
      const float* bzu_i  = pr.bzu + (i - 1) * HDIM;
      const float* wzyu_i = pr.Wzyu + i * HDIM;
#pragma unroll
      for (int j = 0; j < 16; ++j) {
        const int n = (nh * 16 + j) * 16 + col;
        const float bz  = bzu_i[n];
        const float wzy = wzyu_i[n];
#pragma unroll
        for (int e = 0; e < 8; ++e) {
          const int r = rb + e;
          const float t = acc[j][e] + bz + LF(t3c[r]) * wzy;
          LB(Z[r * LDSS + n]) = (bf16_t)fmaxf(p10 * t, 0.0f);
        }
      }
    }
    __syncthreads();

    // GEMM4: new u = relu(10*pu*(u @ Wu[i-1] + bu[i-1])) -> T
    {
      v8f acc[16];
#pragma unroll
      for (int j = 0; j < 16; ++j) acc[j] = (v8f)(0.0f);
      gemm_async_pass(acc, U, pr.ws + (7 + (i - 1)) * HH, pan0, pan1, mt, nh, lane, tid);
      const float* bu_i = pr.bu + (i - 1) * HDIM;
#pragma unroll
      for (int j = 0; j < 16; ++j) {
        const int n = (nh * 16 + j) * 16 + col;
        const float bb = bu_i[n];
#pragma unroll
        for (int e = 0; e < 8; ++e) {
          const int r = rb + e;
          LB(T[r * LDSS + n]) = (bf16_t)fmaxf(pu10 * (acc[j][e] + bb), 0.0f);
        }
      }
    }
    __syncthreads();

    bf16_t* tmp = U; U = T; T = tmp;  // new u becomes U; old U becomes scratch
  }

  // ---- final tail: out = zz.Wzzu_last + u.Wzu_last + bzu_last
  //                        + y*(u.Wyuu[2]+byuu[2])*Wzyu_last ----
  {
    const int r = tid >> 2, q = tid & 3, h0 = q * 128;
    float d1 = 0.0f, d2 = 0.0f, d3 = 0.0f;
    for (int h = h0; h < h0 + 128; ++h) {
      const float uv  = (float)LB(U[r * LDSS + h]);
      const float zzv = (float)LB(T[r * LDSS + h]);
      d1 += zzv * pr.Wzzu_last[h];
      d2 += uv * pr.Wzu_last[h];
      d3 += uv * pr.Wyuu[2 * HDIM + h];
    }
    LF(red[tid]) = d1; LF(red[256 + tid]) = d2; LF(red[512 + tid]) = d3;
  }
  __syncthreads();
  if (tid < 64) {
    const float D1 = LF(red[4*tid]) + LF(red[4*tid+1]) + LF(red[4*tid+2]) + LF(red[4*tid+3]);
    const float D2 = LF(red[256+4*tid]) + LF(red[256+4*tid+1]) + LF(red[256+4*tid+2]) + LF(red[256+4*tid+3]);
    const float D3 = LF(red[512+4*tid]) + LF(red[512+4*tid+1]) + LF(red[512+4*tid+2]) + LF(red[512+4*tid+3]);
    pr.out[gr0 + tid] = D1 + D2 + pr.bzu_last[0]
                      + LF(yv[tid]) * (D3 + pr.byuu[2]) * pr.Wzyu_last[0];
  }
#undef LF
#undef LB
}

// ---------------------------------------------------------------------------
extern "C" void kernel_launch(void* const* d_in, const int* in_sizes, int n_in,
                              void* d_out, int out_size, void* d_ws, size_t ws_size,
                              hipStream_t stream) {
  (void)in_sizes; (void)n_in; (void)out_size; (void)ws_size;

  const float* coords    = (const float*)d_in[0];
  const float* Wu0       = (const float*)d_in[1];
  const float* bu0       = (const float*)d_in[2];
  const float* Wu        = (const float*)d_in[3];
  const float* bu        = (const float*)d_in[4];
  const float* Wzuu      = (const float*)d_in[5];
  const float* bzuu      = (const float*)d_in[6];
  const float* Wzzu      = (const float*)d_in[7];
  const float* Wzzu_last = (const float*)d_in[8];
  const float* Wyuu0     = (const float*)d_in[9];
  const float* byuu0     = (const float*)d_in[10];
  const float* Wyuu      = (const float*)d_in[11];
  const float* byuu      = (const float*)d_in[12];
  const float* Wzyu      = (const float*)d_in[13];
  const float* Wzyu_last = (const float*)d_in[14];
  const float* Wzu0      = (const float*)d_in[15];
  const float* bzu0      = (const float*)d_in[16];
  const float* Wzu       = (const float*)d_in[17];
  const float* bzu       = (const float*)d_in[18];
  const float* Wzu_last  = (const float*)d_in[19];
  const float* bzu_last  = (const float*)d_in[20];
  const float* p         = (const float*)d_in[21];
  const float* pu        = (const float*)d_in[22];
  const float* pzu       = (const float*)d_in[23];

  bf16_t* ws = (bf16_t*)d_ws;

  // prologue: convert+transpose the 9 HxH weight matrices to bf16 in d_ws
  const float* slots[9] = {
    Wzuu, Wzuu + HH, Wzuu + 2 * HH,   // 0,1,2
    Wzzu, Wzzu + HH,                  // 3,4
    Wzu,  Wzu + HH,                   // 5,6
    Wu,   Wu + HH                     // 7,8
  };
  dim3 tb(32, 8), tg(HDIM / 32, HDIM / 32);
  for (int s = 0; s < 9; ++s)
    convert_transpose<<<tg, tb, 0, stream>>>(slots[s], ws + (size_t)s * HH);

  (void)hipFuncSetAttribute((const void*)fcblock_fused,
                            hipFuncAttributeMaxDynamicSharedMemorySize, SMEM_BYTES);

  Params pr;
  pr.coords = coords; pr.Wu0 = Wu0; pr.bu0 = bu0; pr.bu = bu;
  pr.bzuu = bzuu; pr.Wzzu_last = Wzzu_last; pr.Wyuu0 = Wyuu0; pr.byuu0 = byuu0;
  pr.Wyuu = Wyuu; pr.byuu = byuu; pr.Wzyu = Wzyu; pr.Wzyu_last = Wzyu_last;
  pr.Wzu0 = Wzu0; pr.bzu0 = bzu0; pr.bzu = bzu; pr.Wzu_last = Wzu_last;
  pr.bzu_last = bzu_last; pr.p = p; pr.pu = pu; pr.pzu = pzu;
  pr.ws = ws; pr.out = (float*)d_out;

  fcblock_fused<<<NPTS / MROWS, 256, SMEM_BYTES, stream>>>(pr);
}
